// ResidualGatedGCNModel_33741263077799
// MI455X (gfx1250) — compile-verified, matrix operand order checked by reading, not run on hardware
//
#include <hip/hip_runtime.h>
#include <hip/hip_bf16.h>

// ---------------------------------------------------------------------------
// Residual Gated GCN forward for MI455X (gfx1250, wave32, WMMA).
// H=64 GEMMs on v_wmma_f32_16x16x32_bf16. Persistent-tile GEMM: W^T staged to
// LDS once per block, all 8 B fragments pinned in registers, per-tile work is
// a coalesced float2 A stream + 4 ds_load_b128 + 8 WMMAs + branch-free stores.
// Gathers / segment-sum atomics / BN / loss are fp32 float4 streaming.
// ---------------------------------------------------------------------------

typedef __attribute__((ext_vector_type(16))) __bf16 bf16x16;
typedef __attribute__((ext_vector_type(8)))  float  f32x8;

#define HD 64          // hidden size
#define LSTR 72        // LDS row stride in bf16 elems (144 B: 16B-aligned, bank-spread)
#define LWRD 36        // LSTR/2 : 32-bit words per row

// Load a 16-elem bf16 fragment = two contiguous 16B chunks 32B apart.
__device__ __forceinline__ bf16x16 frag_ld(const __bf16* p) {
  union { uint4 q[2]; bf16x16 v; } f;
  f.q[0] = *(const uint4*)(p);
  f.q[1] = *(const uint4*)(p + 16);
  return f.v;
}

__device__ __forceinline__ unsigned pack_bf16(float a, float b) {
  union { __bf16 h[2]; unsigned u; } x;
  x.h[0] = (__bf16)a; x.h[1] = (__bf16)b;
  return x.u;
}

// ---- generic (M,64)@(64,64)+bias GEMM with fused epilogues ---------------
// MODE 0: C = AW + b
// MODE 1: C = relu(AW + b)
// MODE 2: C = AW + b + g1[src[row]] + g1[dst[row]]        (edge update)
// MODE 3: C = AW + b + g1[row] / (g2[row] + 1e-20)        (node update)
template <int MODE>
__global__ __launch_bounds__(128)
void k_gemm64(const float* __restrict__ A, const float* __restrict__ W,
              const float* __restrict__ bias, float* __restrict__ C,
              int M,
              const float* __restrict__ g1, const float* __restrict__ g2,
              const int* __restrict__ srcIdx, const int* __restrict__ dstIdx) {
  __shared__ __align__(16) __bf16 sA[HD * LSTR];   // A tile, row-contiguous
  __shared__ __align__(16) __bf16 sBt[HD * LSTR];  // W transposed: sBt[n*LSTR+k]
  const int tid  = threadIdx.x;
  const int lane = tid & 31;
  const int wave = tid >> 5;
  const int half = lane >> 4;
  const int nl   = lane & 15;
  const int mloc = wave * 16 + nl;

  // --- stage W^T into LDS once per block (L2-resident reads) ---
  unsigned* wB = (unsigned*)sBt;
#pragma unroll
  for (int i = 0; i < 16; ++i) {
    int idx = tid + i * 128;
    int n = idx >> 5, kw = idx & 31, k = 2 * kw;
    wB[n * LWRD + kw] = pack_bf16(W[k * HD + n], W[(k + 1) * HD + n]);
  }
  __syncthreads();

  // --- pin all 8 B fragments in registers for the whole kernel ---
  bf16x16 bf[8];
#pragma unroll
  for (int t = 0; t < 4; ++t) {
    const __bf16* bCol = sBt + (t * 16 + nl) * LSTR + half * 8;
    bf[2 * t]     = frag_ld(bCol);
    bf[2 * t + 1] = frag_ld(bCol + 32);
  }

  float bv[4];
#pragma unroll
  for (int t = 0; t < 4; ++t) bv[t] = bias ? bias[t * 16 + nl] : 0.0f;

  unsigned* wA = (unsigned*)sA;
  const int numTiles = (M + 63) >> 6;
  for (int tile = blockIdx.x; tile < numTiles; tile += gridDim.x) {
    const int rowBase = tile * 64;
    __syncthreads();                       // previous tile's sA reads complete
#pragma unroll
    for (int i = 0; i < 16; ++i) {         // coalesced float2 A stream -> bf16 LDS
      int idx = tid + i * 128;
      int r = idx >> 5, kw = idx & 31;
      int row = rowBase + r;
      float2 av = (row < M) ? *(const float2*)(A + (size_t)row * HD + 2 * kw)
                            : make_float2(0.0f, 0.0f);
      wA[r * LWRD + kw] = pack_bf16(av.x, av.y);
    }
    __syncthreads();

    const __bf16* aRow = sA + mloc * LSTR + half * 8;
    bf16x16 a0 = frag_ld(aRow);            // K 0..31
    bf16x16 a1 = frag_ld(aRow + 32);       // K 32..63

    f32x8 acc[4];
#pragma unroll
    for (int t = 0; t < 4; ++t)
#pragma unroll
      for (int r = 0; r < 8; ++r) acc[t][r] = bv[t];

#pragma unroll
    for (int t = 0; t < 4; ++t) {
      acc[t] = __builtin_amdgcn_wmma_f32_16x16x32_bf16(false, a0, false, bf[2 * t],
                                                       (short)0, acc[t], false, false);
      acc[t] = __builtin_amdgcn_wmma_f32_16x16x32_bf16(false, a1, false, bf[2 * t + 1],
                                                       (short)0, acc[t], false, false);
    }

    if (rowBase + 64 <= M) {
      // ---- full tile: branch-free stores ----
#pragma unroll
      for (int r = 0; r < 8; ++r) {
        const int row = rowBase + wave * 16 + half * 8 + r;
        int si = 0, di = 0;
        if (MODE == 2) { si = srcIdx[row]; di = dstIdx[row]; }
#pragma unroll
        for (int t = 0; t < 4; ++t) {
          const int col = t * 16 + nl;
          float v = acc[t][r];
          if (MODE == 1) v = fmaxf(v, 0.0f);
          else if (MODE == 2)
            v += g1[(size_t)si * HD + col] + g1[(size_t)di * HD + col];
          else if (MODE == 3)
            v += g1[(size_t)row * HD + col] / (g2[(size_t)row * HD + col] + 1e-20f);
          C[(size_t)row * HD + col] = v;
        }
      }
    } else {
      // ---- tail tile: guarded stores ----
#pragma unroll
      for (int r = 0; r < 8; ++r) {
        const int row = rowBase + wave * 16 + half * 8 + r;
        if (row < M) {
          int si = 0, di = 0;
          if (MODE == 2) { si = srcIdx[row]; di = dstIdx[row]; }
#pragma unroll
          for (int t = 0; t < 4; ++t) {
            const int col = t * 16 + nl;
            float v = acc[t][r];
            if (MODE == 1) v = fmaxf(v, 0.0f);
            else if (MODE == 2)
              v += g1[(size_t)si * HD + col] + g1[(size_t)di * HD + col];
            else if (MODE == 3)
              v += g1[(size_t)row * HD + col] / (g2[(size_t)row * HD + col] + 1e-20f);
            C[(size_t)row * HD + col] = v;
          }
        }
      }
    }
  }
}

// ---- small elementwise / reduction kernels --------------------------------
__global__ void k_zero(float* __restrict__ p, size_t n) {
  size_t i = (size_t)blockIdx.x * blockDim.x + threadIdx.x;
  size_t s = (size_t)gridDim.x * blockDim.x;
  for (; i < n; i += s) p[i] = 0.0f;
}

__global__ void k_node_embed(const float* __restrict__ xn, const float* __restrict__ W,
                             float* __restrict__ x, int N) {
  size_t i = (size_t)blockIdx.x * blockDim.x + threadIdx.x;
  if (i >= (size_t)N * HD) return;
  int h = (int)(i & 63);
  size_t node = i >> 6;
  x[i] = xn[node * 2] * W[h] + xn[node * 2 + 1] * W[HD + h];
}

// t = relu(xv * W1 + b1), float4 over columns
__global__ void k_edge_embed1(const float* __restrict__ xv, const float* __restrict__ W1,
                              const float* __restrict__ b1, float* __restrict__ t, int E) {
  size_t i = (size_t)blockIdx.x * blockDim.x + threadIdx.x;
  if (i >= (size_t)E * (HD / 4)) return;
  int c4 = (int)(i & 15);
  size_t eid = i >> 4;
  float v = xv[eid];
  float4 w = ((const float4*)W1)[c4];
  float4 b = ((const float4*)b1)[c4];
  float4 o;
  o.x = fmaxf(v * w.x + b.x, 0.0f);
  o.y = fmaxf(v * w.y + b.y, 0.0f);
  o.z = fmaxf(v * w.z + b.z, 0.0f);
  o.w = fmaxf(v * w.w + b.w, 0.0f);
  ((float4*)t)[i] = o;
}

// gate = sigmoid(e_tmp); segment-sum gate*Vnx[dst] and gate by src
__global__ void k_scatter(const float* __restrict__ e_tmp, const float* __restrict__ Vnx,
                          const int* __restrict__ src, const int* __restrict__ dst,
                          float* __restrict__ agg, float* __restrict__ gsum, int E) {
  size_t i = (size_t)blockIdx.x * blockDim.x + threadIdx.x;
  if (i >= (size_t)E * HD) return;
  int col = (int)(i & 63);
  size_t eid = i >> 6;
  float g = 1.0f / (1.0f + __expf(-e_tmp[i]));
  int s = src[eid], d = dst[eid];
  atomicAdd(&agg[(size_t)s * HD + col], g * Vnx[(size_t)d * HD + col]);
  atomicAdd(&gsum[(size_t)s * HD + col], g);
}

// per-column sum / sumsq over M rows (coalesced: tid&63 = column)
__global__ void k_bn_stats(const float* __restrict__ v, int M, float* __restrict__ stats) {
  int col = threadIdx.x & 63;
  int sub = threadIdx.x >> 6;                       // 0..3
  float s = 0.0f, s2 = 0.0f;
  for (size_t row = (size_t)blockIdx.x * 4 + sub; row < (size_t)M;
       row += (size_t)gridDim.x * 4) {
    float x = v[row * HD + col];
    s += x; s2 += x * x;
  }
  atomicAdd(&stats[col], s);
  atomicAdd(&stats[HD + col], s2);
}

// base += relu( (v - mu) * rsqrt(var + eps) * g + b ), float4 over columns
__global__ void k_bn_apply_res(float* __restrict__ base, const float* __restrict__ v,
                               const float* __restrict__ stats,
                               const float* __restrict__ g, const float* __restrict__ b,
                               int M, float invM) {
  size_t i = (size_t)blockIdx.x * blockDim.x + threadIdx.x;
  if (i >= (size_t)M * (HD / 4)) return;
  int c4 = (int)(i & 15);
  float4 su = ((const float4*)stats)[c4];
  float4 sq = ((const float4*)stats)[16 + c4];
  float4 gg = ((const float4*)g)[c4];
  float4 bb = ((const float4*)b)[c4];
  float4 vv = ((const float4*)v)[i];
  float4 bs = ((const float4*)base)[i];
  float mu, var;
  mu = su.x * invM; var = sq.x * invM - mu * mu;
  bs.x += fmaxf((vv.x - mu) * rsqrtf(var + 1e-5f) * gg.x + bb.x, 0.0f);
  mu = su.y * invM; var = sq.y * invM - mu * mu;
  bs.y += fmaxf((vv.y - mu) * rsqrtf(var + 1e-5f) * gg.y + bb.y, 0.0f);
  mu = su.z * invM; var = sq.z * invM - mu * mu;
  bs.z += fmaxf((vv.z - mu) * rsqrtf(var + 1e-5f) * gg.z + bb.z, 0.0f);
  mu = su.w * invM; var = sq.w * invM - mu * mu;
  bs.w += fmaxf((vv.w - mu) * rsqrtf(var + 1e-5f) * gg.w + bb.w, 0.0f);
  ((float4*)base)[i] = bs;
}

// he = x[src] + x[dst] + e, float4 over columns
__global__ void k_he(const float* __restrict__ x, const float* __restrict__ e,
                     const int* __restrict__ src, const int* __restrict__ dst,
                     float* __restrict__ he, int E) {
  size_t i = (size_t)blockIdx.x * blockDim.x + threadIdx.x;
  if (i >= (size_t)E * (HD / 4)) return;
  int c4 = (int)(i & 15);
  size_t eid = i >> 4;
  int s = src[eid], d = dst[eid];
  float4 xs = ((const float4*)(x + (size_t)s * HD))[c4];
  float4 xd = ((const float4*)(x + (size_t)d * HD))[c4];
  float4 ee = ((const float4*)e)[i];
  float4 o;
  o.x = xs.x + xd.x + ee.x;
  o.y = xs.y + xd.y + ee.y;
  o.z = xs.z + xd.z + ee.z;
  o.w = xs.w + xd.w + ee.w;
  ((float4*)he)[i] = o;
}

// y = he . m3_W + m3_b ; BCE-with-logits, weighted mean into loss_out
__global__ void k_final(const float* __restrict__ he, const float* __restrict__ w3,
                        const float* __restrict__ b3, const int* __restrict__ y_edges,
                        const float* __restrict__ cw, float* __restrict__ y_pred,
                        float* __restrict__ loss_out, int E, float invE) {
  int eid = blockIdx.x * blockDim.x + threadIdx.x;
  if (eid >= E) return;
  float acc = 0.0f;
  const float4* hrow = (const float4*)(he + (size_t)eid * HD);
#pragma unroll
  for (int c = 0; c < HD / 4; ++c) {
    float4 h = hrow[c];
    float4 w = ((const float4*)w3)[c];
    acc += h.x * w.x + h.y * w.y + h.z * w.z + h.w * w.w;
  }
  float y = acc + b3[0];
  y_pred[eid] = y;
  float t = (float)y_edges[eid];
  float ce = fmaxf(y, 0.0f) - y * t + log1pf(__expf(-fabsf(y)));
  atomicAdd(loss_out, cw[0] * ce * invE);
}

// ---------------------------------------------------------------------------
static inline unsigned cdiv(size_t a, size_t b) { return (unsigned)((a + b - 1) / b); }
static inline unsigned minu(unsigned a, unsigned b) { return a < b ? a : b; }

extern "C" void kernel_launch(void* const* d_in, const int* in_sizes, int n_in,
                              void* d_out, int out_size, void* d_ws, size_t ws_size,
                              hipStream_t stream) {
  // ---- inputs (setup_inputs order) ----
  const float* xv        = (const float*)d_in[1];   // (E,1)
  const float* x_nodes   = (const float*)d_in[2];   // (N,2)
  const int*   y_edges   = (const int*)d_in[4];     // (E,)
  const float* edge_cw   = (const float*)d_in[5];   // (1,)
  const int*   edge_idx  = (const int*)d_in[6];     // (2,E)
  const float* nembW     = (const float*)d_in[7];
  const float* ee1W      = (const float*)d_in[8];
  const float* ee1b      = (const float*)d_in[9];
  const float* ee2W      = (const float*)d_in[10];
  const float* ee2b      = (const float*)d_in[11];
  const float* eUW       = (const float*)d_in[12];
  const float* eUb       = (const float*)d_in[13];
  const float* eVW       = (const float*)d_in[14];
  const float* eVb       = (const float*)d_in[15];
  const float* nUW       = (const float*)d_in[16];
  const float* nUb       = (const float*)d_in[17];
  const float* nVW       = (const float*)d_in[18];
  const float* nVb       = (const float*)d_in[19];
  const float* bn_e_g    = (const float*)d_in[20];
  const float* bn_e_b    = (const float*)d_in[21];
  const float* bn_n_g    = (const float*)d_in[22];
  const float* bn_n_b    = (const float*)d_in[23];
  const float* m1W       = (const float*)d_in[24];
  const float* m1b       = (const float*)d_in[25];
  const float* m2W       = (const float*)d_in[26];
  const float* m2b       = (const float*)d_in[27];
  const float* m3W       = (const float*)d_in[28];
  const float* m3b       = (const float*)d_in[29];

  const int E  = in_sizes[0];
  const int N  = in_sizes[2] / 2;
  const int NL = in_sizes[12] / (HD * HD);
  const int* src = edge_idx;
  const int* dst = edge_idx + E;

  // ---- workspace carve ----
  float* p = (float*)d_ws;
  float* x    = p; p += (size_t)N * HD;
  float* e    = p; p += (size_t)E * HD;
  float* etmp = p; p += (size_t)E * HD;   // e_tmp, then he / head activations
  float* Vx   = p; p += (size_t)N * HD;
  float* Vnx  = p; p += (size_t)N * HD;
  float* xt   = p; p += (size_t)N * HD;
  float* agg  = p; p += (size_t)N * HD;
  float* gsum = p; p += (size_t)N * HD;
  float* stats = p; p += 2 * HD;

  float* y_pred = (float*)d_out;
  float* loss   = (float*)d_out + E;

  const dim3 B256(256);
  const unsigned gE  = cdiv((size_t)E * HD, 256);       // scalar elementwise
  const unsigned gE4 = cdiv((size_t)E * (HD / 4), 256); // float4 elementwise
  const unsigned gN4 = cdiv((size_t)N * (HD / 4), 256);
  const unsigned gN  = cdiv((size_t)N * HD, 256);
  const unsigned gmE = minu(cdiv(E, 64), 2048);         // persistent GEMM grids
  const unsigned gmN = minu(cdiv(N, 64), 2048);

  // ---- embeddings ----
  k_node_embed<<<gN, B256, 0, stream>>>(x_nodes, nembW, x, N);
  k_edge_embed1<<<gE4, B256, 0, stream>>>(xv, ee1W, ee1b, etmp, E);
  k_gemm64<0><<<gmE, 128, 0, stream>>>(etmp, ee2W, ee2b, e, E,
                                       nullptr, nullptr, nullptr, nullptr);

  // ---- GCN layers ----
  for (int l = 0; l < NL; ++l) {
    const size_t wOff = (size_t)l * HD * HD;
    const size_t bOff = (size_t)l * HD;
    // Vx = x @ eV + b
    k_gemm64<0><<<gmN, 128, 0, stream>>>(x, eVW + wOff, eVb + bOff, Vx, N,
                                         nullptr, nullptr, nullptr, nullptr);
    // e_tmp = e @ eU + b + Vx[src] + Vx[dst]
    k_gemm64<2><<<gmE, 128, 0, stream>>>(e, eUW + wOff, eUb + bOff, etmp, E,
                                         Vx, nullptr, src, dst);
    // Vnx = x @ nV + b
    k_gemm64<0><<<gmN, 128, 0, stream>>>(x, nVW + wOff, nVb + bOff, Vnx, N,
                                         nullptr, nullptr, nullptr, nullptr);
    // segment sums
    k_zero<<<2048, B256, 0, stream>>>(agg, (size_t)N * HD);
    k_zero<<<2048, B256, 0, stream>>>(gsum, (size_t)N * HD);
    k_scatter<<<gE, B256, 0, stream>>>(etmp, Vnx, src, dst, agg, gsum, E);
    // x_tmp = x @ nU + b + agg/(gsum+1e-20)
    k_gemm64<3><<<gmN, 128, 0, stream>>>(x, nUW + wOff, nUb + bOff, xt, N,
                                         agg, gsum, nullptr, nullptr);
    // e += relu(bn(e_tmp))
    k_zero<<<1, B256, 0, stream>>>(stats, 2 * HD);
    k_bn_stats<<<1024, B256, 0, stream>>>(etmp, E, stats);
    k_bn_apply_res<<<gE4, B256, 0, stream>>>(e, etmp, stats,
                                             bn_e_g + bOff, bn_e_b + bOff, E, 1.0f / E);
    // x += relu(bn(x_tmp))
    k_zero<<<1, B256, 0, stream>>>(stats, 2 * HD);
    k_bn_stats<<<1024, B256, 0, stream>>>(xt, N, stats);
    k_bn_apply_res<<<gN4, B256, 0, stream>>>(x, xt, stats,
                                             bn_n_g + bOff, bn_n_b + bOff, N, 1.0f / N);
  }

  // ---- head MLP over fused per-edge feature ----
  k_he<<<gE4, B256, 0, stream>>>(x, e, src, dst, etmp, E);
  k_gemm64<1><<<gmE, 128, 0, stream>>>(etmp, m1W, m1b, etmp, E,
                                       nullptr, nullptr, nullptr, nullptr);
  k_gemm64<1><<<gmE, 128, 0, stream>>>(etmp, m2W, m2b, etmp, E,
                                       nullptr, nullptr, nullptr, nullptr);
  k_zero<<<1, 32, 0, stream>>>(loss, 1);
  k_final<<<cdiv(E, 256), B256, 0, stream>>>(etmp, m3W, m3b, y_edges, edge_cw,
                                             y_pred, loss, E, 1.0f / E);
}
